// STGCNBlock_42880953483323
// MI455X (gfx1250) — matrix operand order repaired and checked by reading, not compiled, not run on hardware
//
#include <hip/hip_runtime.h>
#include <hip/hip_bf16.h>

typedef _Float16 v16h __attribute__((ext_vector_type(16)));
typedef _Float16 v8h  __attribute__((ext_vector_type(8)));
typedef float    v8f  __attribute__((ext_vector_type(8)));

#define B_  8
#define N_  1024
#define T_  64
#define E_  16384

// ---------------- workspace layout (bytes) ----------------
#define WS_W1F   0            // 5*4*32*16 halves = 20480 B
#define WS_W2F   20480        // 3*4*32*16 halves = 12288 B
#define WS_PF    32768        // 2*32*16 halves   = 2048 B
#define WS_B1    34816        // 64 f32
#define WS_B2    35072        // 64 f32
#define WS_NSUM  35328        // 1024 f32
#define WS_NSQ   39424        // 1024 f32
#define WS_SS    43520        // 2048 f32 (scale,shift per node)
#define WS_H     65536        // B*N*T*16 f32 = 33554432 B
#define WS_T23   (65536 + 33554432)

// ============================================================
// K1: fold 4 conv kernels -> combined tap weights in WMMA
//     B-fragment layout (f16), fold biases, zero BN accums.
//     B layout used by this code (K x 16, contiguous k): lane = (K>=16)*16 + n ; half j -> K = (lane>>4)*16 + j
// ============================================================
struct PrepArgs {
    const float* t1w[4]; const float* t1b[4];
    const float* t2w[4]; const float* t2b[4];
    const float* gcnw;
    _Float16* w1f; _Float16* w2f; _Float16* pf;
    float* b1; float* b2; float* nsum; float* nsq;
};

__global__ __launch_bounds__(128) void stgcn_prep(PrepArgs a) {
    const int gt = blockIdx.x * 128 + threadIdx.x;
    const int ks[4] = {2, 3, 4, 5};
    if (gt < 640) {                       // Weff1 fragments: [d][nt][lane][16]
        int d = gt / 128, r = gt % 128, nt = r / 32, lane = r % 32;
        int lanehi = lane >> 4, o = nt * 16 + (lane & 15);
        for (int j = 0; j < 16; ++j) {
            int K = lanehi * 16 + j;      // channel c = K (Cin=32)
            float s = 0.f;
            for (int i = 0; i < 4; ++i) {
                int k = ks[i];
                if (k - 1 >= d) s += a.t1w[i][(o * 32 + K) * k + (k - 1 - d)];
            }
            a.w1f[((d * 4 + nt) * 32 + lane) * 16 + j] = (_Float16)s;
        }
    } else if (gt < 1024) {               // Weff2 fragments: step s pairs taps (2s, 2s+1)
        int idx = gt - 640;
        int s2 = idx / 128, r = idx % 128, nt = r / 32, lane = r % 32;
        int lanehi = lane >> 4, o = nt * 16 + (lane & 15);
        int d = 2 * s2 + lanehi;
        for (int j = 0; j < 16; ++j) {
            int c = j;                    // K&15
            float s = 0.f;
            if (d <= 4)
                for (int i = 0; i < 4; ++i) {
                    int k = ks[i];
                    if (k - 1 >= d) s += a.t2w[i][(o * 16 + c) * k + (k - 1 - d)];
                }
            a.w2f[((s2 * 4 + nt) * 32 + lane) * 16 + j] = (_Float16)s;
        }
    } else if (gt < 1088) {               // gcn_w (64x16) as 2 K-step B fragments
        int idx = gt - 1024;
        int s2 = idx >> 5, lane = idx & 31, lanehi = lane >> 4, c = lane & 15;
        for (int j = 0; j < 16; ++j) {
            int o = s2 * 32 + lanehi * 16 + j;
            a.pf[(s2 * 32 + lane) * 16 + j] = (_Float16)a.gcnw[o * 16 + c];
        }
    } else if (gt < 1152) {               // summed biases
        int j = gt - 1088;
        float s1 = 0.f, s2 = 0.f;
        for (int i = 0; i < 4; ++i) { s1 += a.t1b[i][j]; s2 += a.t2b[i][j]; }
        a.b1[j] = s1; a.b2[j] = s2;
    } else if (gt < 1152 + 2048) {        // zero BN accumulators
        int j = gt - 1152;
        if (j < 1024) a.nsum[j] = 0.f; else a.nsq[j - 1024] = 0.f;
    }
}

// ============================================================
// K2: temporal conv1 (5-tap, 32->64) + ReLU + 64->16 projection
//     One WG per (b,n); 4 waves, wave w owns M-tile rows 16w..16w+15.
// ============================================================
__global__ __launch_bounds__(128) void stgcn_conv1_proj(
    const float* __restrict__ X, const _Float16* __restrict__ w1f,
    const float* __restrict__ b1, const _Float16* __restrict__ pf,
    float* __restrict__ hout)
{
    __shared__ __align__(16) _Float16 xs[72 * 32];   // rows 0..7 zero pad, row 8+t = X[t]
    __shared__ __align__(32) _Float16 wl[5 * 4 * 32 * 16];
    __shared__ __align__(32) _Float16 pl[2 * 32 * 16];
    __shared__ __align__(16) _Float16 ts[64 * 64];   // relu(t1) tile, f16

    const int tid = threadIdx.x;
    const int bidx = blockIdx.x;                     // b*N + n
    const size_t xbase = (size_t)bidx * (T_ * 32);

    {   // weights -> LDS (u32 copies)
        const uint32_t* s0 = (const uint32_t*)w1f;  uint32_t* d0 = (uint32_t*)wl;
        for (int i = tid; i < 5120; i += 128) d0[i] = s0[i];
        const uint32_t* s1 = (const uint32_t*)pf;   uint32_t* d1 = (uint32_t*)pl;
        for (int i = tid; i < 512; i += 128) d1[i] = s1[i];
    }
    for (int i = tid; i < 8 * 32; i += 128) xs[i] = (_Float16)0.f;
    for (int i = tid; i < T_ * 32; i += 128) xs[8 * 32 + i] = (_Float16)X[xbase + i];
    __syncthreads();

    const int wave = tid >> 5, lane = tid & 31;
    const int lanehi = lane >> 4, lanelo = lane & 15;
    const int tbase = wave * 16;
    const int trow = tbase + lanelo;                 // A-matrix row (lane holds row m=lane&15)
    const int c0 = lanehi * 16;                      // A K-layout channel offset

    v8f acc[4];
    #pragma unroll
    for (int nt = 0; nt < 4; ++nt) {
        float bv = b1[nt * 16 + lanelo];
        #pragma unroll
        for (int i = 0; i < 8; ++i) acc[nt][i] = bv;
    }

    #pragma unroll
    for (int d = 0; d < 5; ++d) {                    // 5 causal taps, K=32 each
        int row = 8 + trow - d;
        v8h lo = *(const v8h*)&xs[row * 32 + c0];        // halves v0..3: K = c0..c0+7
        v8h hi = *(const v8h*)&xs[row * 32 + c0 + 8];    // halves v4..7: K = c0+8..c0+15
        v16h a = __builtin_shufflevector(lo, hi, 0,1,2,3,4,5,6,7,8,9,10,11,12,13,14,15);
        #pragma unroll
        for (int nt = 0; nt < 4; ++nt) {
            v16h bf = *(const v16h*)&wl[((d * 4 + nt) * 32 + lane) * 16];
            acc[nt] = __builtin_amdgcn_wmma_f32_16x16x32_f16(
                false, a, false, bf, (short)0, acc[nt], false, false);
        }
    }

    // ReLU -> f16 tile (C layout: VGPR v -> row tbase+v+8*lanehi, col nt*16+lanelo)
    #pragma unroll
    for (int nt = 0; nt < 4; ++nt)
        #pragma unroll
        for (int v = 0; v < 8; ++v) {
            float r = acc[nt][v]; r = r > 0.f ? r : 0.f;
            ts[(tbase + v + 8 * lanehi) * 64 + nt * 16 + lanelo] = (_Float16)r;
        }
    __syncthreads();

    // projection: h = relu(t1) @ gcn_w   (K=64 -> 2 WMMA steps, one 16-col N-tile)
    v8f accp;
    #pragma unroll
    for (int i = 0; i < 8; ++i) accp[i] = 0.f;
    #pragma unroll
    for (int s = 0; s < 2; ++s) {
        v8h lo = *(const v8h*)&ts[trow * 64 + s * 32 + c0];
        v8h hi = *(const v8h*)&ts[trow * 64 + s * 32 + c0 + 8];
        v16h a = __builtin_shufflevector(lo, hi, 0,1,2,3,4,5,6,7,8,9,10,11,12,13,14,15);
        v16h bf = *(const v16h*)&pl[(s * 32 + lane) * 16];
        accp = __builtin_amdgcn_wmma_f32_16x16x32_f16(
            false, a, false, bf, (short)0, accp, false, false);
    }
    float* hb = hout + (size_t)bidx * (T_ * 16);     // (B,N,T,16), contiguous per (b,n)
    #pragma unroll
    for (int v = 0; v < 8; ++v)
        hb[(tbase + v + 8 * lanehi) * 16 + lanelo] = accp[v];
}

// ============================================================
// K3: GCN edge scatter per (b,t) slice; LDS-resident h + agg.
// ============================================================
__global__ __launch_bounds__(512) void stgcn_scatter(
    const float* __restrict__ h, const int* __restrict__ ei,
    const float* __restrict__ ew, const float* __restrict__ gcnb,
    float* __restrict__ t23)
{
    extern __shared__ float sm[];
    float* hb = sm;                 // N*16 = 16384 f32
    float* ag = sm + N_ * 16;       // N*16
    const int tid = threadIdx.x;
    const int b = blockIdx.x >> 6, t = blockIdx.x & 63;

    for (int i = tid; i < N_ * 16; i += 512) {
        int n = i >> 4, c = i & 15;
        hb[i] = h[(((size_t)(b * N_ + n)) * T_ + t) * 16 + c];
        ag[i] = 0.f;
    }
    __syncthreads();
    for (int e = tid; e < E_; e += 512) {
        int s = ei[e], d = ei[E_ + e];
        float w = ew[e];
        const float* hp = &hb[s * 16];
        float* ap = &ag[d * 16];
        #pragma unroll
        for (int c = 0; c < 16; ++c) atomicAdd(&ap[c], hp[c] * w);  // ds_add_f32
    }
    __syncthreads();
    for (int i = tid; i < N_ * 16; i += 512) {
        int n = i >> 4, c = i & 15;
        float v = ag[i] + gcnb[c];
        t23[(((size_t)(b * N_ + n)) * T_ + t) * 16 + c] = v > 0.f ? v : 0.f;
    }
}

// ============================================================
// K4: temporal conv2 (5-tap, 16->64) + ReLU + BN partial stats.
//     K=32 step s covers taps (2s, 2s+1); tap-5 weights are zero.
// ============================================================
__global__ __launch_bounds__(128) void stgcn_conv2(
    const float* __restrict__ t23, const _Float16* __restrict__ w2f,
    const float* __restrict__ b2, float* __restrict__ out,
    float* __restrict__ nsum, float* __restrict__ nsq)
{
    __shared__ __align__(16) _Float16 xs[72 * 16];
    __shared__ __align__(32) _Float16 wl[3 * 4 * 32 * 16];
    __shared__ float red[128], rq[128];

    const int tid = threadIdx.x;
    const int bidx = blockIdx.x;                     // b*N + n
    const int n = bidx & (N_ - 1);
    const size_t xbase = (size_t)bidx * (T_ * 16);

    {
        const uint32_t* s0 = (const uint32_t*)w2f;  uint32_t* d0 = (uint32_t*)wl;
        for (int i = tid; i < 3072; i += 128) d0[i] = s0[i];
    }
    for (int i = tid; i < 8 * 16; i += 128) xs[i] = (_Float16)0.f;
    for (int i = tid; i < T_ * 16; i += 128) xs[8 * 16 + i] = (_Float16)t23[xbase + i];
    __syncthreads();

    const int wave = tid >> 5, lane = tid & 31;
    const int lanehi = lane >> 4, lanelo = lane & 15;
    const int tbase = wave * 16;
    const int trow = tbase + lanelo;
    const int c0 = lanehi * 8;

    v8f acc[4];
    #pragma unroll
    for (int nt = 0; nt < 4; ++nt) {
        float bv = b2[nt * 16 + lanelo];
        #pragma unroll
        for (int i = 0; i < 8; ++i) acc[nt][i] = bv;
    }

    #pragma unroll
    for (int s = 0; s < 3; ++s) {
        int row = 8 + trow - 2 * s;
        v8h lo = *(const v8h*)&xs[(row - lanehi) * 16];      // K=16h+j: tap d=2s+h, c=j (j<8)
        v8h hi = *(const v8h*)&xs[(row - lanehi) * 16 + 8];  // K=16h+j: tap d=2s+h, c=j (j>=8)
        v16h a = __builtin_shufflevector(lo, hi, 0,1,2,3,4,5,6,7,8,9,10,11,12,13,14,15);
        #pragma unroll
        for (int nt = 0; nt < 4; ++nt) {
            v16h bf = *(const v16h*)&wl[((s * 4 + nt) * 32 + lane) * 16];
            acc[nt] = __builtin_amdgcn_wmma_f32_16x16x32_f16(
                false, a, false, bf, (short)0, acc[nt], false, false);
        }
    }

    float psum = 0.f, psq = 0.f;
    float* ob = out + (size_t)bidx * (T_ * 64);
    #pragma unroll
    for (int nt = 0; nt < 4; ++nt)
        #pragma unroll
        for (int v = 0; v < 8; ++v) {
            float r = acc[nt][v]; r = r > 0.f ? r : 0.f;
            ob[(tbase + v + 8 * lanehi) * 64 + nt * 16 + lanelo] = r;
            psum += r; psq += r * r;
        }

    red[tid] = psum; rq[tid] = psq;
    __syncthreads();
    for (int off = 64; off > 0; off >>= 1) {
        if (tid < off) { red[tid] += red[tid + off]; rq[tid] += rq[tid + off]; }
        __syncthreads();
    }
    if (tid == 0) {
        atomicAdd(&nsum[n], red[0]);
        atomicAdd(&nsq[n], rq[0]);
    }
}

// ============================================================
// K5: per-node scale/shift   K6: in-place normalize (float4)
// ============================================================
__global__ __launch_bounds__(256) void stgcn_bn_final(
    const float* __restrict__ nsum, const float* __restrict__ nsq,
    const float* __restrict__ gamma, const float* __restrict__ beta,
    float* __restrict__ ss)
{
    int n = blockIdx.x * 256 + threadIdx.x;
    if (n < N_) {
        const float inv = 1.f / (float)(B_ * T_ * 64);
        float m = nsum[n] * inv;
        float v = nsq[n] * inv - m * m;
        float sc = gamma[n] * rsqrtf(v + 1e-5f);
        ss[2 * n] = sc;
        ss[2 * n + 1] = beta[n] - m * sc;
    }
}

__global__ __launch_bounds__(256) void stgcn_bn_apply(
    float* __restrict__ out, const float* __restrict__ ss)
{
    size_t i = (size_t)blockIdx.x * 256 + threadIdx.x;   // float4 index
    int n = (int)((i >> 10) & (N_ - 1));                 // 4096 elems per (b,n) => 1024 float4
    float sc = ss[2 * n], sh = ss[2 * n + 1];
    float4* p = (float4*)out;
    float4 v = p[i];
    v.x = v.x * sc + sh; v.y = v.y * sc + sh;
    v.z = v.z * sc + sh; v.w = v.w * sc + sh;
    p[i] = v;
}

// ============================================================
extern "C" void kernel_launch(void* const* d_in, const int* in_sizes, int n_in,
                              void* d_out, int out_size, void* d_ws, size_t ws_size,
                              hipStream_t stream) {
    const float* X    = (const float*)d_in[0];
    const int*   ei   = (const int*)d_in[2];
    const float* ew   = (const float*)d_in[3];
    const float* gcnw = (const float*)d_in[12];
    const float* gcnb = (const float*)d_in[13];
    const float* gamma = (const float*)d_in[22];
    const float* beta  = (const float*)d_in[23];

    char* ws = (char*)d_ws;
    _Float16* w1f = (_Float16*)(ws + WS_W1F);
    _Float16* w2f = (_Float16*)(ws + WS_W2F);
    _Float16* pf  = (_Float16*)(ws + WS_PF);
    float* b1   = (float*)(ws + WS_B1);
    float* b2   = (float*)(ws + WS_B2);
    float* nsum = (float*)(ws + WS_NSUM);
    float* nsq  = (float*)(ws + WS_NSQ);
    float* ssb  = (float*)(ws + WS_SS);
    float* hbuf = (float*)(ws + WS_H);
    float* t23  = (float*)(ws + WS_T23);
    float* out  = (float*)d_out;

    PrepArgs pa;
    for (int i = 0; i < 4; ++i) {
        pa.t1w[i] = (const float*)d_in[4 + i];
        pa.t1b[i] = (const float*)d_in[8 + i];
        pa.t2w[i] = (const float*)d_in[14 + i];
        pa.t2b[i] = (const float*)d_in[18 + i];
    }
    pa.gcnw = gcnw;
    pa.w1f = w1f; pa.w2f = w2f; pa.pf = pf;
    pa.b1 = b1; pa.b2 = b2; pa.nsum = nsum; pa.nsq = nsq;

    stgcn_prep<<<25, 128, 0, stream>>>(pa);
    stgcn_conv1_proj<<<B_ * N_, 128, 0, stream>>>(X, w1f, b1, pf, hbuf);
    stgcn_scatter<<<B_ * T_, 512, 2 * N_ * 16 * sizeof(float), stream>>>(hbuf, ei, ew, gcnb, t23);
    stgcn_conv2<<<B_ * N_, 128, 0, stream>>>(t23, w2f, b2, out, nsum, nsq);
    stgcn_bn_final<<<4, 256, 0, stream>>>(nsum, nsq, gamma, beta, ssb);
    stgcn_bn_apply<<<(B_ * N_ * T_ * 64) / (256 * 4), 256, 0, stream>>>(out, ssb);
}
